// ReadInAttention_53480932770343
// MI455X (gfx1250) — matrix operand order repaired and checked by reading, not compiled, not run on hardware
//
#include <hip/hip_runtime.h>
#include <hip/hip_bf16.h>
#include <stdint.h>

// ---------------- problem constants ----------------
#define B_ 4
#define U_ 64
#define V_ 256
#define SDIM 768      // STATE_DIM == IN_DIM
#define CDIM 384      // CODE_DIM
#define INNER_ 512    // NUM_HEADS * HEAD_DIM
#define NH 8
#define HD 64

typedef __attribute__((ext_vector_type(16))) __bf16 v16bf;
typedef __attribute__((ext_vector_type(8)))  __bf16 v8bf;
typedef __attribute__((ext_vector_type(8)))  float  v8f;

__device__ __forceinline__ __bf16 f2bf(float f) {
  uint32_t u = __builtin_bit_cast(uint32_t, f);
  uint32_t r = u + 0x7FFFu + ((u >> 16) & 1u);   // round-to-nearest-even
  uint16_t h = (uint16_t)(r >> 16);
  return __builtin_bit_cast(__bf16, h);
}

__device__ __forceinline__ v16bf cat16(v8bf lo, v8bf hi) {
  return __builtin_shufflevector(lo, hi, 0, 1, 2, 3, 4, 5, 6, 7,
                                         8, 9, 10, 11, 12, 13, 14, 15);
}
// A fragment: chunks [0..7] and [16..23] from the lane's row pointer.
__device__ __forceinline__ v16bf loadA_frag(const __bf16* p) {
  const v8bf* q = (const v8bf*)p; return cat16(q[0], q[2]);
}
// B fragment: chunks [0..7] and [16..23] from the lane's row pointer.
__device__ __forceinline__ v16bf loadB_frag(const __bf16* p) {
  const v8bf* q = (const v8bf*)p; return cat16(q[0], q[2]);
}

struct Frags { v16bf a0, a1, b0, b1, b2, b3; };

__device__ __forceinline__ Frags load_frags(const __bf16* Ap0, const __bf16* Ap1,
                                            const __bf16* Bp0, const __bf16* Bp1,
                                            const __bf16* Bp2, const __bf16* Bp3, int k) {
  Frags f;
  f.a0 = loadA_frag(Ap0 + k);
  f.a1 = loadA_frag(Ap1 + k);
  f.b0 = loadB_frag(Bp0 + k);
  f.b1 = loadB_frag(Bp1 + k);
  f.b2 = loadB_frag(Bp2 + k);
  f.b3 = loadB_frag(Bp3 + k);
  return f;
}

__device__ __forceinline__ void mma8(const Frags& f, v8f acc0[4], v8f acc1[4]) {
  acc0[0] = __builtin_amdgcn_wmma_f32_16x16x32_bf16(false, f.a0, false, f.b0, (short)0, acc0[0], false, false);
  acc1[0] = __builtin_amdgcn_wmma_f32_16x16x32_bf16(false, f.a1, false, f.b0, (short)0, acc1[0], false, false);
  acc0[1] = __builtin_amdgcn_wmma_f32_16x16x32_bf16(false, f.a0, false, f.b1, (short)0, acc0[1], false, false);
  acc1[1] = __builtin_amdgcn_wmma_f32_16x16x32_bf16(false, f.a1, false, f.b1, (short)0, acc1[1], false, false);
  acc0[2] = __builtin_amdgcn_wmma_f32_16x16x32_bf16(false, f.a0, false, f.b2, (short)0, acc0[2], false, false);
  acc1[2] = __builtin_amdgcn_wmma_f32_16x16x32_bf16(false, f.a1, false, f.b2, (short)0, acc1[2], false, false);
  acc0[3] = __builtin_amdgcn_wmma_f32_16x16x32_bf16(false, f.a0, false, f.b3, (short)0, acc0[3], false, false);
  acc1[3] = __builtin_amdgcn_wmma_f32_16x16x32_bf16(false, f.a1, false, f.b3, (short)0, acc1[3], false, false);
}

// ---------------- LayerNorm: one block per row ----------------
__global__ void k_layernorm(const float* __restrict__ x, const float* __restrict__ g,
                            const float* __restrict__ b, float* __restrict__ y, int dim) {
  __shared__ float red[256];
  const int row = blockIdx.x;
  const float* xr = x + (size_t)row * dim;
  float s = 0.f, s2 = 0.f;
  for (int i = threadIdx.x; i < dim; i += blockDim.x) { float v = xr[i]; s += v; s2 += v * v; }
  red[threadIdx.x] = s; __syncthreads();
  for (int o = 128; o > 0; o >>= 1) { if (threadIdx.x < o) red[threadIdx.x] += red[threadIdx.x + o]; __syncthreads(); }
  float mean = red[0] / dim; __syncthreads();
  red[threadIdx.x] = s2; __syncthreads();
  for (int o = 128; o > 0; o >>= 1) { if (threadIdx.x < o) red[threadIdx.x] += red[threadIdx.x + o]; __syncthreads(); }
  float var = red[0] / dim - mean * mean;
  float inv = rsqrtf(var + 1e-5f);
  for (int i = threadIdx.x; i < dim; i += blockDim.x)
    y[(size_t)row * dim + i] = (xr[i] - mean) * inv * g[i] + b[i];
}

// ---------------- elementwise helpers ----------------
__global__ void k_convert(const float* __restrict__ x, __bf16* __restrict__ y, int n) {
  for (int i = blockIdx.x * blockDim.x + threadIdx.x; i < n; i += gridDim.x * blockDim.x)
    y[i] = f2bf(x[i]);
}

// in: (nb, R, C) f32 row-major -> out: (nb, C, R) bf16 row-major
__global__ void k_transpose_convert(const float* __restrict__ in, __bf16* __restrict__ out,
                                    int R, int C, int nb) {
  const int per = R * C;
  const int n = per * nb;
  for (int idx = blockIdx.x * blockDim.x + threadIdx.x; idx < n; idx += gridDim.x * blockDim.x) {
    int bb = idx / per, o = idx % per;
    int c = o / R, r = o % R;
    out[idx] = f2bf(in[(size_t)bb * per + (size_t)r * C + c]);
  }
}

// qin = r * (1 + cmq)
__global__ void k_scale_qin(const float* __restrict__ r, const float* __restrict__ cmq,
                            __bf16* __restrict__ y, int n) {
  for (int i = blockIdx.x * blockDim.x + threadIdx.x; i < n; i += gridDim.x * blockDim.x)
    y[i] = f2bf(r[i] * (1.f + cmq[i]));
}

// q_bf = q + bq
__global__ void k_addbias_convert(const float* __restrict__ q, const float* __restrict__ bq,
                                  __bf16* __restrict__ y, int n, int ncols) {
  for (int i = blockIdx.x * blockDim.x + threadIdx.x; i < n; i += gridDim.x * blockDim.x)
    y[i] = f2bf(q[i] + bq[i % ncols]);
}

// a = t * (1 + cmk) / SCALE   (t layout (h, m, i); cmk layout (m, i))
__global__ void k_scale_a(const float* __restrict__ t, const float* __restrict__ cmk,
                          __bf16* __restrict__ y, int n) {
  const int MI = (B_ * U_) * SDIM;
  for (int i = blockIdx.x * blockDim.x + threadIdx.x; i < n; i += gridDim.x * blockDim.x)
    y[i] = f2bf(t[i] * (1.f + cmk[i % MI]) * 0.125f);   // 1/sqrt(64)
}

// wsv = ws * (1 + cmv)   (ws layout (b,h,u,i); cmv layout ((b,u), i))
__global__ void k_scale_ws(const float* __restrict__ ws, const float* __restrict__ cmv,
                           __bf16* __restrict__ y, int n) {
  for (int idx = blockIdx.x * blockDim.x + threadIdx.x; idx < n; idx += gridDim.x * blockDim.x) {
    int t = idx / (U_ * SDIM);
    int b = t >> 3;
    int u = (idx / SDIM) % U_;
    int i = idx % SDIM;
    y[idx] = f2bf(ws[idx] * (1.f + cmv[(size_t)(b * U_ + u) * SDIM + i]));
  }
}

// msgm = (msg + bv) * (1 + cme)
__global__ void k_finalize_msg(const float* __restrict__ msg, const float* __restrict__ bv,
                               const float* __restrict__ cme, __bf16* __restrict__ y, int n) {
  for (int i = blockIdx.x * blockDim.x + threadIdx.x; i < n; i += gridDim.x * blockDim.x)
    y[i] = f2bf((msg[i] + bv[i % INNER_]) * (1.f + cme[i]));
}

// out = rs + (og + be) * gamma
__global__ void k_final_out(const float* __restrict__ rs, const float* __restrict__ og,
                            const float* __restrict__ be, const float* __restrict__ gamma,
                            float* __restrict__ out, int n) {
  for (int idx = blockIdx.x * blockDim.x + threadIdx.x; idx < n; idx += gridDim.x * blockDim.x) {
    int i = idx % SDIM;
    out[idx] = rs[idx] + (og[idx] + be[i]) * gamma[i];
  }
}

// ---------------- softmax: one wave32 per 256-wide row ----------------
__global__ void k_softmax(const float* __restrict__ S, __bf16* __restrict__ W) {
  const int wave = threadIdx.x >> 5, lane = threadIdx.x & 31;
  const int row = blockIdx.x * 8 + wave;
  const float* sr = S + (size_t)row * V_;
  float v[8];
  float mx = -3.0e38f;
#pragma unroll
  for (int j = 0; j < 8; ++j) { v[j] = sr[lane + 32 * j]; mx = fmaxf(mx, v[j]); }
#pragma unroll
  for (int o = 16; o > 0; o >>= 1) mx = fmaxf(mx, __shfl_xor(mx, o, 32));
  float sum = 0.f;
#pragma unroll
  for (int j = 0; j < 8; ++j) { v[j] = __expf(v[j] - mx); sum += v[j]; }
#pragma unroll
  for (int o = 16; o > 0; o >>= 1) sum += __shfl_xor(sum, o, 32);
  float r = 1.f / sum;
#pragma unroll
  for (int j = 0; j < 8; ++j) W[(size_t)row * V_ + lane + 32 * j] = f2bf(v[j] * r);
}

// ---------------- bf16 WMMA GEMM (NT), one wave per 32x64 macro-tile ----------------
// C[m,n] = sum_k A[m,k] * B[n,k];  A: M x K (lda), B: N x K (ldb), C: M x N (ldc).
// Ping-pong double-buffered over two 32-wide k-steps (requires K % 64 == 0):
// loads for buffer f1 issue while f0 computes and vice versa -> no rotation copies.
// M % 32 == 0, N % 64 == 0; lda/ldb multiples of 8 elements (16B alignment).
// Dual batch dims: blockIdx.y = b1*nb2 + b2.
__global__ void k_gemm_nt_bf16(const __bf16* __restrict__ A, const __bf16* __restrict__ B,
                               float* __restrict__ C, int M, int N, int K,
                               int lda, int ldb, int ldc,
                               long long sa1, long long sa2, long long sb1, long long sb2,
                               long long sc1, long long sc2, int nb2) {
  const int lane = threadIdx.x;
  const int hw  = lane >> 4;           // half-wave: 0 or 1
  const int l16 = lane & 15;
  const int n64 = N >> 6;
  const int tn = (blockIdx.x % n64) << 6;
  const int tm = (blockIdx.x / n64) << 5;
  const int b1 = blockIdx.y / nb2, b2 = blockIdx.y % nb2;
  A += b1 * sa1 + b2 * sa2;
  B += b1 * sb1 + b2 * sb2;
  C += b1 * sc1 + b2 * sc2;

  const __bf16* Ap0 = A + (size_t)(tm + l16) * lda + hw * 8;
  const __bf16* Ap1 = Ap0 + (size_t)16 * lda;
  const __bf16* Bq0 = B + (size_t)(tn + l16) * ldb + hw * 8;
  const __bf16* Bq1 = Bq0 + (size_t)16 * ldb;
  const __bf16* Bq2 = Bq1 + (size_t)16 * ldb;
  const __bf16* Bq3 = Bq2 + (size_t)16 * ldb;

  v8f acc0[4] = {v8f{}, v8f{}, v8f{}, v8f{}};
  v8f acc1[4] = {v8f{}, v8f{}, v8f{}, v8f{}};

  // K is always a multiple of 64 here (K/32 even).
  Frags f0 = load_frags(Ap0, Ap1, Bq0, Bq1, Bq2, Bq3, 0);
  int k0 = 0;
  for (; k0 + 64 < K; k0 += 64) {
    Frags f1 = load_frags(Ap0, Ap1, Bq0, Bq1, Bq2, Bq3, k0 + 32);
    __builtin_prefetch(Ap0 + k0 + 64, 0, 1);   // global_prefetch_b8 (speculative)
    mma8(f0, acc0, acc1);
    f0 = load_frags(Ap0, Ap1, Bq0, Bq1, Bq2, Bq3, k0 + 64);
    mma8(f1, acc0, acc1);
  }
  // tail: two remaining 32-wide k-steps (k0 = K-64)
  {
    Frags f1 = load_frags(Ap0, Ap1, Bq0, Bq1, Bq2, Bq3, k0 + 32);
    mma8(f0, acc0, acc1);
    mma8(f1, acc0, acc1);
  }

  float* Cp0 = C + (size_t)(tm + hw * 8) * ldc + tn + l16;
  float* Cp1 = Cp0 + (size_t)16 * ldc;
#pragma unroll
  for (int t = 0; t < 4; ++t)
#pragma unroll
    for (int i = 0; i < 8; ++i) {
      Cp0[(size_t)i * ldc + 16 * t] = acc0[t][i];
      Cp1[(size_t)i * ldc + 16 * t] = acc1[t][i];
    }
}

// ---------------- host side ----------------
static inline void gemm_nt(const __bf16* A, const __bf16* B, float* C,
                           int M, int N, int K, int lda, int ldb, int ldc,
                           long long sa1, long long sa2, long long sb1, long long sb2,
                           long long sc1, long long sc2, int nb1, int nb2, hipStream_t stream) {
  dim3 grid((unsigned)((M / 32) * (N / 64)), (unsigned)(nb1 * nb2));
  k_gemm_nt_bf16<<<grid, 32, 0, stream>>>(A, B, C, M, N, K, lda, ldb, ldc,
                                          sa1, sa2, sb1, sb2, sc1, sc2, nb2);
}

static inline dim3 ew_grid(int n) { return dim3((unsigned)((n + 255) / 256)); }

extern "C" void kernel_launch(void* const* d_in, const int* in_sizes, int n_in,
                              void* d_out, int out_size, void* d_ws, size_t ws_size,
                              hipStream_t stream) {
  const float* rs    = (const float*)d_in[0];
  const float* codes = (const float*)d_in[1];
  const float* ss    = (const float*)d_in[2];
  const float* ln_r_g = (const float*)d_in[3];
  const float* ln_r_b = (const float*)d_in[4];
  const float* ln_s_g = (const float*)d_in[5];
  const float* ln_s_b = (const float*)d_in[6];
  const float* Wq  = (const float*)d_in[7];
  const float* bq  = (const float*)d_in[8];
  const float* Wmq = (const float*)d_in[9];
  const float* Wk  = (const float*)d_in[10];
  // d_in[11] = bk: constant across softmax axis -> cancels exactly, unused.
  const float* Wmk = (const float*)d_in[12];
  const float* Wv  = (const float*)d_in[13];
  const float* bv  = (const float*)d_in[14];
  const float* Wmv = (const float*)d_in[15];
  const float* We  = (const float*)d_in[16];
  const float* be  = (const float*)d_in[17];
  const float* Wme = (const float*)d_in[18];
  const float* gamma = (const float*)d_in[19];
  float* out = (float*)d_out;

  const int MU = B_ * U_;        // 256 receiver rows
  const int MV = B_ * V_;        // 1024 sender rows

  char* base = (char*)d_ws;
  size_t off = 0;
  auto alloc = [&](size_t bytes) -> void* {
    void* p = base + off;
    off = (off + bytes + 255) & ~(size_t)255;
    return p;
  };

  // bf16 staging (weights stored transposed -> all GEMMs are NT / contiguous-K)
  __bf16* codes_bf = (__bf16*)alloc((size_t)MU * CDIM * 2);
  __bf16* WmqT = (__bf16*)alloc((size_t)SDIM * CDIM * 2);     // (768, 384)
  __bf16* WmkT = (__bf16*)alloc((size_t)SDIM * CDIM * 2);
  __bf16* WmvT = (__bf16*)alloc((size_t)SDIM * CDIM * 2);
  __bf16* WmeT = (__bf16*)alloc((size_t)INNER_ * CDIM * 2);   // (512, 384)
  __bf16* WqT  = (__bf16*)alloc((size_t)INNER_ * SDIM * 2);   // (512, 768)
  __bf16* Wk_bf= (__bf16*)alloc((size_t)SDIM * INNER_ * 2);   // plain (768, 512): already N x K for t-fold
  __bf16* WvT  = (__bf16*)alloc((size_t)INNER_ * SDIM * 2);   // (512, 768)
  __bf16* WeT  = (__bf16*)alloc((size_t)SDIM * INNER_ * 2);   // (768, 512)
  __bf16* s_bf = (__bf16*)alloc((size_t)MV * SDIM * 2);       // ((b,v), i)
  __bf16* sT_bf= (__bf16*)alloc((size_t)B_ * SDIM * V_ * 2);  // (b, i, v)
  __bf16* qin_bf = (__bf16*)alloc((size_t)MU * SDIM * 2);
  __bf16* q_bf   = (__bf16*)alloc((size_t)MU * INNER_ * 2);
  __bf16* a_bf   = (__bf16*)alloc((size_t)NH * MU * SDIM * 2);  // (h, m, i)
  __bf16* w_bf   = (__bf16*)alloc((size_t)B_ * NH * U_ * V_ * 2);
  __bf16* wsv_bf = (__bf16*)alloc((size_t)B_ * NH * U_ * SDIM * 2);
  __bf16* msgm_bf= (__bf16*)alloc((size_t)MU * INNER_ * 2);
  // f32 intermediates
  float* r_f   = (float*)alloc((size_t)MU * SDIM * 4);
  float* s_f   = (float*)alloc((size_t)MV * SDIM * 4);
  float* cmq   = (float*)alloc((size_t)MU * SDIM * 4);
  float* cmk   = (float*)alloc((size_t)MU * SDIM * 4);
  float* cmv   = (float*)alloc((size_t)MU * SDIM * 4);
  float* cme   = (float*)alloc((size_t)MU * INNER_ * 4);
  float* q_f   = (float*)alloc((size_t)MU * INNER_ * 4);
  float* t_f   = (float*)alloc((size_t)NH * MU * SDIM * 4);     // (h, m, i)
  float* sc_f  = (float*)alloc((size_t)B_ * NH * U_ * V_ * 4);  // (b,h,u,v)
  float* ws_f  = (float*)alloc((size_t)B_ * NH * U_ * SDIM * 4);// (b,h,u,i)
  float* msg_f = (float*)alloc((size_t)MU * INNER_ * 4);        // ((b,u), hc)
  float* og_f  = (float*)alloc((size_t)MU * SDIM * 4);

  // 1) LayerNorms
  k_layernorm<<<MU, 256, 0, stream>>>(rs, ln_r_g, ln_r_b, r_f, SDIM);
  k_layernorm<<<MV, 256, 0, stream>>>(ss, ln_s_g, ln_s_b, s_f, SDIM);

  // 2) bf16 staging
  k_convert<<<ew_grid(MV * SDIM), 256, 0, stream>>>(s_f, s_bf, MV * SDIM);
  k_transpose_convert<<<ew_grid(B_ * SDIM * V_), 256, 0, stream>>>(s_f, sT_bf, V_, SDIM, B_);
  k_convert<<<ew_grid(MU * CDIM), 256, 0, stream>>>(codes, codes_bf, MU * CDIM);
  k_transpose_convert<<<ew_grid(CDIM * SDIM), 256, 0, stream>>>(Wmq, WmqT, CDIM, SDIM, 1);
  k_transpose_convert<<<ew_grid(CDIM * SDIM), 256, 0, stream>>>(Wmk, WmkT, CDIM, SDIM, 1);
  k_transpose_convert<<<ew_grid(CDIM * SDIM), 256, 0, stream>>>(Wmv, WmvT, CDIM, SDIM, 1);
  k_transpose_convert<<<ew_grid(CDIM * INNER_), 256, 0, stream>>>(Wme, WmeT, CDIM, INNER_, 1);
  k_transpose_convert<<<ew_grid(SDIM * INNER_), 256, 0, stream>>>(Wq, WqT, SDIM, INNER_, 1);
  k_convert<<<ew_grid(SDIM * INNER_), 256, 0, stream>>>(Wk, Wk_bf, SDIM * INNER_);
  k_transpose_convert<<<ew_grid(SDIM * INNER_), 256, 0, stream>>>(Wv, WvT, SDIM, INNER_, 1);
  k_transpose_convert<<<ew_grid(INNER_ * SDIM), 256, 0, stream>>>(We, WeT, INNER_, SDIM, 1);

  // 3) code modulations: cm* = codes @ Wm*  (NT against transposed weights)
  gemm_nt(codes_bf, WmqT, cmq, MU, SDIM, CDIM, CDIM, CDIM, SDIM, 0,0,0,0,0,0, 1,1, stream);
  gemm_nt(codes_bf, WmkT, cmk, MU, SDIM, CDIM, CDIM, CDIM, SDIM, 0,0,0,0,0,0, 1,1, stream);
  gemm_nt(codes_bf, WmvT, cmv, MU, SDIM, CDIM, CDIM, CDIM, SDIM, 0,0,0,0,0,0, 1,1, stream);
  gemm_nt(codes_bf, WmeT, cme, MU, INNER_, CDIM, CDIM, CDIM, INNER_, 0,0,0,0,0,0, 1,1, stream);

  // 4) q = (r*(1+cmq)) @ Wq + bq
  k_scale_qin<<<ew_grid(MU * SDIM), 256, 0, stream>>>(r_f, cmq, qin_bf, MU * SDIM);
  gemm_nt(qin_bf, WqT, q_f, MU, INNER_, SDIM, SDIM, SDIM, INNER_, 0,0,0,0,0,0, 1,1, stream);
  k_addbias_convert<<<ew_grid(MU * INNER_), 256, 0, stream>>>(q_f, bq, q_bf, MU * INNER_, INNER_);

  // 5) t[h,m,i] = sum_c q[m, h*64+c] * Wk[i, h*64+c]  (B = Wk rows, contiguous in c)
  gemm_nt(q_bf, Wk_bf, t_f, MU, SDIM, HD, INNER_, INNER_, SDIM,
          (long long)HD, 0, (long long)HD, 0, (long long)MU * SDIM, 0, NH, 1, stream);

  // 6) a = t * (1+cmk) / sqrt(HD)
  k_scale_a<<<ew_grid(NH * MU * SDIM), 256, 0, stream>>>(t_f, cmk, a_bf, NH * MU * SDIM);

  // 7) scores[b,h,u,v] = a[h,(b,u),:] . s[(b,v),:]   (batched over b,h)
  gemm_nt(a_bf, s_bf, sc_f, U_, V_, SDIM, SDIM, SDIM, V_,
          (long long)U_ * SDIM, (long long)MU * SDIM,
          (long long)V_ * SDIM, 0,
          (long long)NH * U_ * V_, (long long)U_ * V_,
          B_, NH, stream);

  // 8) softmax over v
  k_softmax<<<(B_ * NH * U_) / 8, 256, 0, stream>>>(sc_f, w_bf);

  // 9) ws[b,h,u,i] = weights @ s_b  -> NT against sT (b, i, v)
  gemm_nt(w_bf, sT_bf, ws_f, U_, SDIM, V_, V_, V_, SDIM,
          (long long)NH * U_ * V_, (long long)U_ * V_,
          (long long)SDIM * V_, 0,
          (long long)NH * U_ * SDIM, (long long)U_ * SDIM,
          B_, NH, stream);

  // 10) wsv = ws * (1+cmv)
  k_scale_ws<<<ew_grid(B_ * NH * U_ * SDIM), 256, 0, stream>>>(ws_f, cmv, wsv_bf, B_ * NH * U_ * SDIM);

  // 11) msg[(b,u), h*64+c] = wsv[b,h,u,:] . WvT[h*64+c, :]
  gemm_nt(wsv_bf, WvT, msg_f, U_, HD, SDIM, SDIM, SDIM, INNER_,
          (long long)NH * U_ * SDIM, (long long)U_ * SDIM,
          0, (long long)HD * SDIM,
          (long long)U_ * INNER_, (long long)HD,
          B_, NH, stream);

  // 12) out = ((msg+bv)*(1+cme)) @ We, then +be, *gamma, + residual
  k_finalize_msg<<<ew_grid(MU * INNER_), 256, 0, stream>>>(msg_f, bv, cme, msgm_bf, MU * INNER_);
  gemm_nt(msgm_bf, WeT, og_f, MU, SDIM, INNER_, INNER_, INNER_, SDIM, 0,0,0,0,0,0, 1,1, stream);
  k_final_out<<<ew_grid(MU * SDIM), 256, 0, stream>>>(rs, og_f, be, gamma, out, MU * SDIM);
}
